// CustomConvWithExtra_31533649887369
// MI455X (gfx1250) — matrix operand order repaired and checked by reading, not compiled, not run on hardware
//
#include <hip/hip_runtime.h>

typedef __attribute__((ext_vector_type(16))) __bf16 v16bf;
typedef __attribute__((ext_vector_type(8)))  __bf16 v8bf;
typedef __attribute__((ext_vector_type(8)))  float  v8f;
typedef __attribute__((ext_vector_type(4)))  float  v4f;

#define BATCH 32
#define CIN   64
#define COUT  64
#define FES   3
#define HH    128
#define WW    128

#define TY 8
#define TX 32
#define SLAB_ROWS 10
#define SLAB_COLS 34
#define PIXSTRIDE 72          // bf16 elems per pixel slot (64 used, pad to 16B mult)
#define OUT_ROWSTRIDE 36      // floats
#define OUT_COSTRIDE  292     // floats (8*36+4 -> avoids bank alignment pathologies)
#define SMEM_BYTES 74752      // max(slab: 340*72*2=48960, outbuf: 64*292*4=74752)

#define WFRAG_BYTES (9*2*4*32*32)   // 73728 B of bf16 B-fragments

__device__ __forceinline__ unsigned short f2bf(float f) {
    unsigned u = __float_as_uint(f);
    u += 0x7fffu + ((u >> 16) & 1u);     // round-to-nearest-even
    return (unsigned short)(u >> 16);
}

// ---------------------------------------------------------------------------
// Pre-kernel 1: repack conv_w[64][64][3][3] f32 -> bf16 WMMA B-fragments.
// Layout: [tap(9)][kc(2)][nt(4)][lane(32)][e(16)] ushort.
// B (32x16 bf16): lane holds N = lane&15; K = e + 16*(lane>>4).
// ---------------------------------------------------------------------------
__global__ void repack_weights(const float* __restrict__ conv_w,
                               unsigned short* __restrict__ wfrag) {
    int t = blockIdx.x * blockDim.x + threadIdx.x;
    if (t >= 9*2*4*32*16) return;
    int e    = t & 15;
    int lane = (t >> 4) & 31;
    int nt   = (t >> 9) & 3;
    int kc   = (t >> 11) & 1;
    int tap  = t >> 12;
    int ki = tap / 3, kj = tap % 3;
    int K  = e + ((lane >> 4) << 4);
    int ci = kc * 32 + K;
    int co = nt * 16 + (lane & 15);
    wfrag[t] = f2bf(conv_w[((co * CIN + ci) * 3 + ki) * 3 + kj]);
}

// ---------------------------------------------------------------------------
// Pre-kernel 2: fold spatially-constant "extra" path + both biases into a
// per-(b, cout, border-case) table. 9 cases: (top/mid/bot) x (left/mid/right).
// ---------------------------------------------------------------------------
__global__ void build_ttab(const float* __restrict__ extra_inputs,
                           const float* __restrict__ conv_b,
                           const float* __restrict__ extra_w,
                           const float* __restrict__ extra_b,
                           float* __restrict__ ttab) {
    int t = blockIdx.x * blockDim.x + threadIdx.x;
    if (t >= BATCH * COUT * 9) return;
    int cse = t % 9;
    int co  = (t / 9) % COUT;
    int b   = t / (9 * COUT);
    int cy = cse / 3, cx = cse % 3;
    int ki0 = (cy == 0) ? 1 : 0, ki1 = (cy == 2) ? 1 : 2;
    int kj0 = (cx == 0) ? 1 : 0, kj1 = (cx == 2) ? 1 : 2;
    float acc = conv_b[co] + extra_b[co];
    for (int f = 0; f < FES; ++f) {
        float s = 0.f;
        for (int ki = ki0; ki <= ki1; ++ki)
            for (int kj = kj0; kj <= kj1; ++kj)
                s += extra_w[((co * FES + f) * 3 + ki) * 3 + kj];
        acc += extra_inputs[b * (COUT * FES) + co * FES + f] * s;
    }
    ttab[t] = acc;   // index = (b*COUT + co)*9 + cse
}

// ---------------------------------------------------------------------------
// Main kernel: implicit-GEMM 3x3 conv via v_wmma_f32_16x16x32_bf16.
// Block = 256 threads (8 waves). Tile: 1 batch, 8 rows x 32 cols, all 64 cout.
// Wave (nt = w&3, g = w>>2) owns cout tile nt and x-half g; 8 row accumulators.
// Tap loop kept rolled (unroll 1) to bound live fragments -> no VGPR spills.
// ---------------------------------------------------------------------------
__global__ __launch_bounds__(256, 4) void conv_main(
        const float* __restrict__ xin,
        const unsigned short* __restrict__ wfrag,
        const float* __restrict__ ttab,
        float* __restrict__ out) {
    __shared__ __align__(16) unsigned char smem[SMEM_BYTES];
    unsigned short* slab = (unsigned short*)smem;      // phase A
    float* outbuf = (float*)smem;                      // phase B (after barrier)

    const int tid = threadIdx.x;
    const int bid = blockIdx.x;
    const int xb = bid & 3;
    const int yb = (bid >> 2) & 15;
    const int b  = bid >> 6;
    const int x0 = xb * TX;
    const int y0 = yb * TY;

    // ---- stage 1: global f32 -> LDS bf16 slab, [pixel][cin], zero-padded halo
    const float* xbase = xin + (size_t)b * CIN * HH * WW;
    #pragma unroll 1
    for (int i = 0; i < 85; ++i) {                 // 85*256 == 10*34*64
        int li   = tid + i * 256;
        int c    = li % SLAB_COLS;
        int rest = li / SLAB_COLS;
        int ci   = rest & 63;
        int r    = rest >> 6;
        int gy = y0 - 1 + r;
        int gx = x0 - 1 + c;
        float v = 0.f;
        if ((unsigned)gy < 128u && (unsigned)gx < 128u)
            v = xbase[(ci * HH + gy) * WW + gx];
        slab[(r * SLAB_COLS + c) * PIXSTRIDE + ci] = f2bf(v);
    }
    __syncthreads();

    // ---- stage 2: WMMA main loop
    const int lane = tid & 31;
    const int w    = tid >> 5;
    const int nt   = w & 3;
    const int g    = w >> 2;
    const int m    = lane & 15;
    const int lh   = lane >> 4;

    v8f acc[8];
    #pragma unroll
    for (int i = 0; i < 8; ++i)
        acc[i] = (v8f){0.f,0.f,0.f,0.f,0.f,0.f,0.f,0.f};

    const v8bf* wb = (const v8bf*)wfrag;
    #pragma unroll 1
    for (int tap = 0; tap < 9; ++tap) {
        const int di = tap / 3 - 1;
        const int dj = tap - (di + 1) * 3 - 1;
        const int cidx = g * 16 + m + dj + 1;             // slab col, in [0,33]
        const unsigned short* arow =
            slab + ((di + 1) * SLAB_COLS + cidx) * PIXSTRIDE + lh * 16;
        #pragma unroll
        for (int kc = 0; kc < 2; ++kc) {
            const int widx = ((((tap * 2 + kc) * 4 + nt) * 32) + lane) * 2;
            v8bf blo = wb[widx];
            v8bf bhi = wb[widx + 1];
            v16bf bf = __builtin_shufflevector(blo, bhi,
                0,1,2,3,4,5,6,7,8,9,10,11,12,13,14,15);
            const unsigned short* ap0 = arow + kc * 32;
            #pragma unroll
            for (int i = 0; i < 8; ++i) {
                const unsigned short* ap = ap0 + i * (SLAB_COLS * PIXSTRIDE);
                v8bf alo = *(const v8bf*)ap;             // K: base+0..7
                v8bf ahi = *(const v8bf*)(ap + 8);       // K: base+8..15
                v16bf af = __builtin_shufflevector(alo, ahi,
                    0,1,2,3,4,5,6,7,8,9,10,11,12,13,14,15);
                acc[i] = __builtin_amdgcn_wmma_f32_16x16x32_bf16(
                    false, af, false, bf, (short)0, acc[i], false, false);
            }
        }
    }
    __syncthreads();   // slab dead; smem becomes outbuf

    // ---- stage 3: C fragments -> LDS transpose buffer [cout][row][x]
    const int xloc = g * 16 + lh * 8;
    const int col  = nt * 16 + m;
    #pragma unroll
    for (int i = 0; i < 8; ++i) {
        float* p = outbuf + col * OUT_COSTRIDE + i * OUT_ROWSTRIDE + xloc;
        *(v4f*)p       = __builtin_shufflevector(acc[i], acc[i], 0,1,2,3);
        *(v4f*)(p + 4) = __builtin_shufflevector(acc[i], acc[i], 4,5,6,7);
    }
    __syncthreads();

    // ---- stage 4: fused epilogue (+T table) and coalesced b128 stores
    const int co = tid >> 2;
    const int q  = tid & 3;
    const float* tp = ttab + (b * COUT + co) * 9;
    const int gx0 = x0 + q * 8;
    float* obase = out + ((size_t)(b * COUT + co) * HH + y0) * WW + gx0;
    #pragma unroll
    for (int i = 0; i < 8; ++i) {
        const int gy = y0 + i;
        const int cy = (gy == 0) ? 0 : ((gy == 127) ? 2 : 1);
        const float t0 = tp[cy * 3 + 0];
        const float t1 = tp[cy * 3 + 1];
        const float t2 = tp[cy * 3 + 2];
        const float* lp = outbuf + co * OUT_COSTRIDE + i * OUT_ROWSTRIDE + q * 8;
        v4f lo = *(const v4f*)lp;
        v4f hi = *(const v4f*)(lp + 4);
        const float addL = (gx0 == 0) ? t0 : t1;          // x==0 only at j==0
        const float addR = (gx0 + 7 == 127) ? t2 : t1;    // x==127 only at j==7
        v4f aL = {addL, t1, t1, t1};
        v4f aR = {t1, t1, t1, addR};
        float* po = obase + i * WW;
        *(v4f*)po       = lo + aL;
        *(v4f*)(po + 4) = hi + aR;
    }
}

// ---------------------------------------------------------------------------
extern "C" void kernel_launch(void* const* d_in, const int* in_sizes, int n_in,
                              void* d_out, int out_size, void* d_ws, size_t ws_size,
                              hipStream_t stream) {
    const float* x  = (const float*)d_in[0];
    const float* ei = (const float*)d_in[1];
    const float* cw = (const float*)d_in[2];
    const float* cb = (const float*)d_in[3];
    const float* ew = (const float*)d_in[4];
    const float* eb = (const float*)d_in[5];
    float* out = (float*)d_out;

    unsigned short* wfrag = (unsigned short*)d_ws;
    float* ttab = (float*)((char*)d_ws + WFRAG_BYTES);

    repack_weights<<<(9*2*4*32*16 + 255) / 256, 256, 0, stream>>>(cw, wfrag);
    build_ttab<<<(BATCH*COUT*9 + 255) / 256, 256, 0, stream>>>(ei, cb, ew, eb, ttab);
    conv_main<<<BATCH * (HH/TY) * (WW/TX), 256, 0, stream>>>(x, wfrag, ttab, out);
}
